// SMoG_73023033966956
// MI455X (gfx1250) — matrix operand order repaired and checked by reading, not compiled, not run on hardware
//
#include <hip/hip_runtime.h>
#include <hip/hip_bf16.h>
#include <math.h>

typedef __attribute__((ext_vector_type(16))) _Float16 v16h;
typedef __attribute__((ext_vector_type(8)))  _Float16 v8h;
typedef __attribute__((ext_vector_type(4)))  _Float16 v4h;
typedef __attribute__((ext_vector_type(8)))  float    v8f;

#define B_SAMPLES 16384
#define G_GROUPS  8192
#define DIM       1024
#define BETA      0.99f
#define EPSN      1e-12f

#define M_TILE 32              // rows of x per workgroup (2 row-tiles/wave)
#define WAVES  8               // 8 waves -> 128 columns per slab
#define COLS_PER_ITER (WAVES * 16)
#define LDS_STRIDE 1032        // halves per LDS row: 1024 + 8 pad (bank-conflict-free)

// ---------------------------------------------------------------------------
// Phase 1: per-group inverse L2 norm, f32 -> f16 conversion of g, zero counts
// ---------------------------------------------------------------------------
__global__ void prep_g_kernel(const float* __restrict__ g,
                              _Float16* __restrict__ gh,
                              float* __restrict__ invn,
                              int* __restrict__ counts) {
    const int j = blockIdx.x;           // group row
    const int t = threadIdx.x;          // 256 threads, 4 elems each
    const float4 gv = ((const float4*)(g + (size_t)j * DIM))[t];

    v4h h;
    h[0] = (_Float16)gv.x; h[1] = (_Float16)gv.y;
    h[2] = (_Float16)gv.z; h[3] = (_Float16)gv.w;
    ((v4h*)(gh + (size_t)j * DIM))[t] = h;

    float s = gv.x * gv.x + gv.y * gv.y + gv.z * gv.z + gv.w * gv.w;
    #pragma unroll
    for (int m = 16; m > 0; m >>= 1) s += __shfl_xor(s, m, 32);

    __shared__ float wsum[8];
    const int lane = t & 31, wid = t >> 5;
    if (lane == 0) wsum[wid] = s;
    __syncthreads();
    if (t == 0) {
        float tot = 0.f;
        #pragma unroll
        for (int w = 0; w < 8; ++w) tot += wsum[w];
        invn[j] = 1.0f / fmaxf(sqrtf(tot), EPSN);
        counts[j] = 0;
    }
}

// ---------------------------------------------------------------------------
// Phase 2: fused  S = x @ g^T  (f16 WMMA, f32 accum), scale cols by 1/||g||,
// running argmax per row. Writes assignments + atomic counts.
// ---------------------------------------------------------------------------
__global__ void __launch_bounds__(256)
gemm_argmax_kernel(const float* __restrict__ x,
                   const _Float16* __restrict__ gh,
                   const float* __restrict__ invn,
                   int* __restrict__ assign,
                   int* __restrict__ counts) {
    __shared__ _Float16 xs[M_TILE * LDS_STRIDE];   // 66 KB, padded
    __shared__ float redV[WAVES * 32];
    __shared__ int   redI[WAVES * 32];

    const int tid     = threadIdx.x;
    const int lane    = tid & 31;
    const int wid     = tid >> 5;
    const int rowBase = blockIdx.x * M_TILE;

    // Stage 32 rows of x into LDS as f16 (converted on the fly; x read once).
    for (int c = tid; c < M_TILE * DIM / 4; c += 256) {
        const int row = c >> 8;           // 256 float4 chunks per row
        const int cp  = c & 255;
        const float4 xv = ((const float4*)(x + (size_t)(rowBase + row) * DIM))[cp];
        v4h h;
        h[0] = (_Float16)xv.x; h[1] = (_Float16)xv.y;
        h[2] = (_Float16)xv.z; h[3] = (_Float16)xv.w;
        *((v4h*)&xs[row * LDS_STRIDE + cp * 4]) = h;
    }
    __syncthreads();

    const int half = lane >> 4;     // 0 or 1
    const int l15  = lane & 15;

    float bestV0[8], bestV1[8];
    int   bestI0[8], bestI1[8];
    #pragma unroll
    for (int r = 0; r < 8; ++r) {
        bestV0[r] = -__builtin_inff(); bestV1[r] = -__builtin_inff();
        bestI0[r] = 0;                 bestI1[r] = 0;
    }

    // A fragment base: row = l15 (tile0) / l15+16 (tile1); K lo at 8*half, hi +16
    const _Float16* a0base = &xs[l15 * LDS_STRIDE + half * 8];
    const _Float16* a1base = a0base + 16 * LDS_STRIDE;

    for (int it = 0; it < G_GROUPS / COLS_PER_ITER; ++it) {
        const int col = it * COLS_PER_ITER + wid * 16 + l15;
        // B fragment: column = l15; K lo at 8*half, hi +16 (matches A)
        const _Float16* gp = gh + (size_t)col * DIM + half * 8;

        v8f c0 = {}; v8f c1 = {};
        #pragma unroll 4
        for (int k0 = 0; k0 < DIM; k0 += 32) {
            const v8h blo = *((const v8h*)(gp + k0));
            const v8h bhi = *((const v8h*)(gp + k0 + 16));
            const v16h b = __builtin_shufflevector(blo, bhi,
                0,1,2,3,4,5,6,7,8,9,10,11,12,13,14,15);

            const v8h a0lo = *((const v8h*)(a0base + k0));
            const v8h a0hi = *((const v8h*)(a0base + k0 + 16));
            const v16h a0 = __builtin_shufflevector(a0lo, a0hi,
                0,1,2,3,4,5,6,7,8,9,10,11,12,13,14,15);
            const v8h a1lo = *((const v8h*)(a1base + k0));
            const v8h a1hi = *((const v8h*)(a1base + k0 + 16));
            const v16h a1 = __builtin_shufflevector(a1lo, a1hi,
                0,1,2,3,4,5,6,7,8,9,10,11,12,13,14,15);

            c0 = __builtin_amdgcn_wmma_f32_16x16x32_f16(
                false, a0, false, b, (short)0, c0, false, false);
            c1 = __builtin_amdgcn_wmma_f32_16x16x32_f16(
                false, a1, false, b, (short)0, c1, false, false);
        }

        const float inv = invn[col];   // this lane's column scale (1/||g_col||)
        #pragma unroll
        for (int r = 0; r < 8; ++r) {
            const float v0 = c0[r] * inv;
            if (v0 > bestV0[r]) { bestV0[r] = v0; bestI0[r] = col; }
            const float v1 = c1[r] * inv;
            if (v1 > bestV1[r]) { bestV1[r] = v1; bestI1[r] = col; }
        }
    }

    // Cross-lane argmax: row M lives in 16 lanes of a half-wave (C layout:
    // VGPR r -> M = r + 8*half, N = lane&15). Tie-break: lowest group index.
    #pragma unroll
    for (int r = 0; r < 8; ++r) {
        float v0 = bestV0[r]; int i0 = bestI0[r];
        float v1 = bestV1[r]; int i1 = bestI1[r];
        #pragma unroll
        for (int m = 1; m < 16; m <<= 1) {
            float ov = __shfl_xor(v0, m, 32); int oi = __shfl_xor(i0, m, 32);
            if (ov > v0 || (ov == v0 && oi < i0)) { v0 = ov; i0 = oi; }
            ov = __shfl_xor(v1, m, 32); oi = __shfl_xor(i1, m, 32);
            if (ov > v1 || (ov == v1 && oi < i1)) { v1 = ov; i1 = oi; }
        }
        if (l15 == 0) {
            const int row0 = 8 * half + r;          // 0..15 (tile0)
            redV[wid * 32 + row0]      = v0; redI[wid * 32 + row0]      = i0;
            redV[wid * 32 + 16 + row0] = v1; redI[wid * 32 + 16 + row0] = i1;
        }
    }
    __syncthreads();

    if (tid < 32) {                 // one thread per row of this row-block
        float bv = redV[tid]; int bi = redI[tid];
        #pragma unroll
        for (int w = 1; w < WAVES; ++w) {
            const float v = redV[w * 32 + tid]; const int i = redI[w * 32 + tid];
            if (v > bv || (v == bv && i < bi)) { bv = v; bi = i; }
        }
        assign[rowBase + tid] = bi;
        atomicAdd(&counts[bi], 1);
    }
}

// ---------------------------------------------------------------------------
// Phase 3: acc = g * (counts>0 ? beta : 1)
// ---------------------------------------------------------------------------
__global__ void init_acc_kernel(const float* __restrict__ g,
                                const int* __restrict__ counts,
                                float* __restrict__ acc) {
    const int j = blockIdx.x;
    const int t = threadIdx.x;
    const float f = (counts[j] > 0) ? BETA : 1.0f;
    float4 gv = ((const float4*)(g + (size_t)j * DIM))[t];
    gv.x *= f; gv.y *= f; gv.z *= f; gv.w *= f;
    ((float4*)(acc + (size_t)j * DIM))[t] = gv;
}

// ---------------------------------------------------------------------------
// Phase 4: acc[assign[i]] += (1-beta) * x[i] / counts[assign[i]]
// ---------------------------------------------------------------------------
__global__ void scatter_kernel(const float* __restrict__ x,
                               const int* __restrict__ assign,
                               const int* __restrict__ counts,
                               float* __restrict__ acc) {
    const int i = blockIdx.x;
    const int t = threadIdx.x;
    const int a = assign[i];
    const float scale = (1.0f - BETA) / (float)counts[a];
    const float4 xv = ((const float4*)(x + (size_t)i * DIM))[t];
    float* p = acc + (size_t)a * DIM + t * 4;
    atomicAdd(p + 0, scale * xv.x);
    atomicAdd(p + 1, scale * xv.y);
    atomicAdd(p + 2, scale * xv.z);
    atomicAdd(p + 3, scale * xv.w);
}

// ---------------------------------------------------------------------------
// Phase 5: out = normalize(acc) rows
// ---------------------------------------------------------------------------
__global__ void finalize_kernel(const float* __restrict__ acc,
                                float* __restrict__ out) {
    const int j = blockIdx.x;
    const int t = threadIdx.x;
    float4 av = ((const float4*)(acc + (size_t)j * DIM))[t];
    float s = av.x * av.x + av.y * av.y + av.z * av.z + av.w * av.w;
    #pragma unroll
    for (int m = 16; m > 0; m >>= 1) s += __shfl_xor(s, m, 32);

    __shared__ float wsum[8];
    __shared__ float sinv;
    const int lane = t & 31, wid = t >> 5;
    if (lane == 0) wsum[wid] = s;
    __syncthreads();
    if (t == 0) {
        float tot = 0.f;
        #pragma unroll
        for (int w = 0; w < 8; ++w) tot += wsum[w];
        sinv = 1.0f / fmaxf(sqrtf(tot), EPSN);
    }
    __syncthreads();
    const float inv = sinv;
    av.x *= inv; av.y *= inv; av.z *= inv; av.w *= inv;
    ((float4*)(out + (size_t)j * DIM))[t] = av;
}

// ---------------------------------------------------------------------------
extern "C" void kernel_launch(void* const* d_in, const int* in_sizes, int n_in,
                              void* d_out, int out_size, void* d_ws, size_t ws_size,
                              hipStream_t stream) {
    const float* x = (const float*)d_in[0];          // [16384, 1024]
    const float* g = (const float*)d_in[1];          // [8192, 1024]
    float* out = (float*)d_out;                      // [8192, 1024]

    char* ws = (char*)d_ws;
    _Float16* gh  = (_Float16*)ws;  ws += (size_t)G_GROUPS * DIM * sizeof(_Float16);
    float*    invn = (float*)ws;    ws += (size_t)G_GROUPS * sizeof(float);
    int*      assign = (int*)ws;    ws += (size_t)B_SAMPLES * sizeof(int);
    int*      counts = (int*)ws;    ws += (size_t)G_GROUPS * sizeof(int);
    float*    acc = (float*)ws;     // [8192, 1024] f32

    prep_g_kernel<<<G_GROUPS, 256, 0, stream>>>(g, gh, invn, counts);
    gemm_argmax_kernel<<<B_SAMPLES / M_TILE, 256, 0, stream>>>(x, gh, invn, assign, counts);
    init_acc_kernel<<<G_GROUPS, 256, 0, stream>>>(g, counts, acc);
    scatter_kernel<<<B_SAMPLES, 256, 0, stream>>>(x, assign, counts, acc);
    finalize_kernel<<<G_GROUPS, 256, 0, stream>>>(acc, out);
}
